// HAN_84688165143117
// MI455X (gfx1250) — matrix operand order repaired and checked by reading, not compiled, not run on hardware
//
#include <hip/hip_runtime.h>
#include <hip/hip_bf16.h>

typedef __attribute__((ext_vector_type(16))) _Float16 v16h;
typedef __attribute__((ext_vector_type(8)))  _Float16 v8h;
typedef __attribute__((ext_vector_type(8)))  float    v8f;

#define HID     128
#define HEADS   8
#define DHEAD   16
#define NNODE   10000
#define NE      640000
#define NP      1000000
#define LAYERS  2
#define NEG_SLOPE 0.2f
#define MROWS   80        // rows per block = 5 m-tiles; 10000 = 125 * 80
#define APAD    136       // A row stride in halves (272B = 17*16B, keeps 16B align)

// ---------------------------------------------------------------------------
// WMMA GEMM: C[M,128] = A[M,128] @ W[128,128] + bias, optional relu.
// Block = 256 threads (8 waves) x 80 rows. W staged once per block into LDS,
// pre-swizzled into B-fragment layout; A tile staged as f16 rows. Each wave
// owns one 16-col tile and loops 5 m-tiles x 4 K-steps of
// v_wmma_f32_16x16x32_f16 with ds_load_b128 fragment reads.
// ---------------------------------------------------------------------------
__global__ __launch_bounds__(256) void han_gemm_kernel(
    const float* __restrict__ A, const float* __restrict__ W,
    const float* __restrict__ bias, float* __restrict__ C, int relu)
{
    __shared__ _Float16 Al[MROWS * APAD];          // 21.25 KB
    __shared__ _Float16 Wl[4 * 8 * 2 * 32 * 8];    // 32 KB, fragment-swizzled

    // ---- stage W (128x128 f32 -> f16, swizzled to B-fragment layout) ----
    // frag layout: [kt][ntile][chunk][lane][8 halves]; lane = half*16 + (n&15)
    #pragma unroll 4
    for (int it = 0; it < 64; ++it) {
        const int idx = it * 256 + threadIdx.x;    // coalesced global read
        const int k = idx >> 7, n = idx & 127;
        const int kt = k >> 5, hf = (k >> 4) & 1, e = k & 15;
        const int c = e >> 3, eo = e & 7;
        const int nt = n >> 4, ll = n & 15;
        Wl[(((kt * 8 + nt) * 2 + c) * 32 + (hf * 16 + ll)) * 8 + eo] = (_Float16)W[idx];
    }
    // ---- stage A tile (80x128 f32 -> f16, row-major padded) ----
    #pragma unroll 4
    for (int it = 0; it < 40; ++it) {
        const int idx = it * 256 + threadIdx.x;
        const int r = idx >> 7, k = idx & 127;
        Al[r * APAD + k] = (_Float16)A[((size_t)blockIdx.x * MROWS + r) * HID + k];
    }
    __syncthreads();

    const int wave = threadIdx.x >> 5;             // n-tile 0..7
    const int lane = threadIdx.x & 31;
    const int half = lane >> 4;
    const int l    = lane & 15;
    const int ncol = wave * 16 + l;
    const float bb = bias[ncol];

    for (int mt = 0; mt < 5; ++mt) {
        v8f acc = {};
        #pragma unroll
        for (int kt = 0; kt < 4; ++kt) {
            const int k0 = kt * 32;
            // A 16x32 f16 fragment: lane row l, contiguous K 16*half..16*half+15
            v8h a0 = *(const v8h*)&Al[(mt * 16 + l) * APAD + k0 + half * 16];
            v8h a1 = *(const v8h*)&Al[(mt * 16 + l) * APAD + k0 + half * 16 + 8];
            v16h a = __builtin_shufflevector(a0, a1,
                        0,1,2,3,4,5,6,7,8,9,10,11,12,13,14,15);
            // B 32x16 f16 fragment: two contiguous 16B chunks per lane
            v8h b0 = *(const v8h*)&Wl[(((kt * 8 + wave) * 2 + 0) * 32 + lane) * 8];
            v8h b1 = *(const v8h*)&Wl[(((kt * 8 + wave) * 2 + 1) * 32 + lane) * 8];
            v16h b = __builtin_shufflevector(b0, b1,
                        0,1,2,3,4,5,6,7,8,9,10,11,12,13,14,15);
            acc = __builtin_amdgcn_wmma_f32_16x16x32_f16(
                      false, a, false, b, (short)0, acc, false, false);
        }
        #pragma unroll
        for (int r = 0; r < 8; ++r) {
            const int orow = blockIdx.x * MROWS + mt * 16 + half * 8 + r;
            float v = acc[r] + bb;
            if (relu) v = fmaxf(v, 0.f);
            C[(size_t)orow * HID + ncol] = v;
        }
    }
}

// s[node,h] = sum_d p[node,h,d] * a[h,d]
__global__ void han_score_kernel(const float* __restrict__ p,
                                 const float* __restrict__ a,
                                 float* __restrict__ s)
{
    int i = blockIdx.x * blockDim.x + threadIdx.x;       // node*HEADS + h
    if (i >= NNODE * HEADS) return;
    const int node = i >> 3, h = i & 7;
    const float* pv = p + node * HID + h * DHEAD;
    const float* av = a + h * DHEAD;
    float acc = 0.f;
    #pragma unroll
    for (int d = 0; d < DHEAD; ++d) acc += pv[d] * av[d];
    s[i] = acc;
}

__global__ void han_fill_u32(unsigned* __restrict__ p, unsigned v, int n)
{
    int i = blockIdx.x * blockDim.x + threadIdx.x;
    if (i < n) p[i] = v;
}

__device__ __forceinline__ unsigned han_ord_from_f(float f) {
    unsigned u = __float_as_uint(f);
    return (u & 0x80000000u) ? ~u : (u | 0x80000000u);
}
__device__ __forceinline__ float han_f_from_ord(unsigned u) {
    unsigned b = (u & 0x80000000u) ? (u ^ 0x80000000u) : ~u;
    return __uint_as_float(b);
}
__device__ __forceinline__ float han_leaky(float a) {
    return a >= 0.f ? a : NEG_SLOPE * a;
}

// pass 1: segment max of leaky(alpha) over dst, via ordered-uint atomicMax
__global__ void han_edge_max_kernel(const int* __restrict__ ei,
                                    const float* __restrict__ ssrc,
                                    const float* __restrict__ sdst,
                                    unsigned* __restrict__ amax_u)
{
    int e = blockIdx.x * blockDim.x + threadIdx.x;
    if (e >= NE) return;
    const int src = ei[e], dst = ei[NE + e];
    #pragma unroll
    for (int h = 0; h < HEADS; ++h) {
        float al = han_leaky(ssrc[src * HEADS + h] + sdst[dst * HEADS + h]);
        atomicMax(&amax_u[dst * HEADS + h], han_ord_from_f(al));
    }
}

__global__ void han_amax_fin_kernel(const unsigned* __restrict__ amax_u,
                                    float* __restrict__ amax_f)
{
    int i = blockIdx.x * blockDim.x + threadIdx.x;
    if (i >= NNODE * HEADS) return;
    float v = han_f_from_ord(amax_u[i]);
    if (!isfinite(v)) v = 0.f;      // segments with no incoming edges
    amax_f[i] = v;
}

// pass 2: den[dst,h] += exp(alpha - amax[dst,h])
__global__ void han_edge_den_kernel(const int* __restrict__ ei,
                                    const float* __restrict__ ssrc,
                                    const float* __restrict__ sdst,
                                    const float* __restrict__ amax_f,
                                    float* __restrict__ den)
{
    int e = blockIdx.x * blockDim.x + threadIdx.x;
    if (e >= NE) return;
    const int src = ei[e], dst = ei[NE + e];
    #pragma unroll
    for (int h = 0; h < HEADS; ++h) {
        float al = han_leaky(ssrc[src * HEADS + h] + sdst[dst * HEADS + h]);
        atomicAdd(&den[dst * HEADS + h], __expf(al - amax_f[dst * HEADS + h]));
    }
}

// pass 3: agg[dst,:] += p_src[src,:] * coef ; one wave per edge, float4 per lane
__global__ __launch_bounds__(256) void han_edge_scatter_kernel(
    const int* __restrict__ ei, const float* __restrict__ psrc,
    const float* __restrict__ ssrc, const float* __restrict__ sdst,
    const float* __restrict__ amax_f, const float* __restrict__ den,
    float* __restrict__ agg)
{
    const int wave = threadIdx.x >> 5, lane = threadIdx.x & 31;
    const int e = blockIdx.x * 8 + wave;               // NE % 8 == 0
    if (e >= NE) return;
    const int src = ei[e], dst = ei[NE + e];
    const int h = lane >> 2;                           // 4 lanes per head
    float al  = han_leaky(ssrc[src * HEADS + h] + sdst[dst * HEADS + h]);
    float ex  = __expf(al - amax_f[dst * HEADS + h]);
    float coef = ex / fmaxf(den[dst * HEADS + h], 1e-16f);
    const int off = lane * 4;
    const float4 pv = *(const float4*)(psrc + (size_t)src * HID + off);
    float* ob = agg + (size_t)dst * HID + off;
    atomicAdd(ob + 0, pv.x * coef);
    atomicAdd(ob + 1, pv.y * coef);
    atomicAdd(ob + 2, pv.z * coef);
    atomicAdd(ob + 3, pv.w * coef);
}

// relu + write into next-layer h and into the concatenated output table slice
__global__ void han_relu_copy_kernel(const float* __restrict__ agg,
                                     float* __restrict__ hout,
                                     float* __restrict__ Etab, int layer)
{
    int i = blockIdx.x * blockDim.x + threadIdx.x;
    if (i >= NNODE * HID) return;
    const int node = i >> 7, col = i & 127;
    float v = fmaxf(agg[i], 0.f);
    hout[i] = v;
    Etab[(size_t)node * (HID * LAYERS) + layer * HID + col] = v;
}

// one wave per pair: out[p] = dot(Em[m], Ed[d]) over 256 floats
__global__ __launch_bounds__(256) void han_pair_kernel(
    const int* __restrict__ pidx, const float* __restrict__ Em,
    const float* __restrict__ Ed, float* __restrict__ out)
{
    const int wave = threadIdx.x >> 5, lane = threadIdx.x & 31;
    const int p = blockIdx.x * 8 + wave;               // NP % 8 == 0
    if (p >= NP) return;
    const int m = pidx[p], d = pidx[NP + p];
    const float4* em = (const float4*)(Em + (size_t)m * 256);
    const float4* ed = (const float4*)(Ed + (size_t)d * 256);
    float acc = 0.f;
    #pragma unroll
    for (int q = 0; q < 2; ++q) {
        float4 a = em[lane * 2 + q];
        float4 b = ed[lane * 2 + q];
        acc += a.x * b.x + a.y * b.y + a.z * b.z + a.w * b.w;
    }
    #pragma unroll
    for (int o = 16; o > 0; o >>= 1) acc += __shfl_xor(acc, o, 32);
    if (lane == 0) out[p] = acc;
}

// ---------------------------------------------------------------------------
extern "C" void kernel_launch(void* const* d_in, const int* in_sizes, int n_in,
                              void* d_out, int out_size, void* d_ws, size_t ws_size,
                              hipStream_t stream)
{
    const float* x1     = (const float*)d_in[0];
    const float* x2     = (const float*)d_in[1];
    const float* lin1_w = (const float*)d_in[2];
    const float* lin1_b = (const float*)d_in[3];
    const float* lin2_w = (const float*)d_in[4];
    const float* lin2_b = (const float*)d_in[5];
    const int*   e12    = (const int*)d_in[22];
    const int*   e21    = (const int*)d_in[23];
    const int*   pidx   = (const int*)d_in[24];
    float* out = (float*)d_out;

    // workspace carve-up (floats); all offsets 16B-aligned
    float* ws = (float*)d_ws;
    size_t o = 0;
    auto alloc = [&](size_t n) { float* p = ws + o; o += n; return p; };
    float* h1   = alloc((size_t)NNODE * HID);
    float* h2   = alloc((size_t)NNODE * HID);
    float* p1   = alloc((size_t)NNODE * HID);
    float* p2   = alloc((size_t)NNODE * HID);
    float* agg  = alloc((size_t)NNODE * HID);
    float* sA   = alloc((size_t)NNODE * HEADS);
    float* sB   = alloc((size_t)NNODE * HEADS);
    float* amaxF= alloc((size_t)NNODE * HEADS);
    float* den  = alloc((size_t)NNODE * HEADS);
    unsigned* amaxU = (unsigned*)alloc((size_t)NNODE * HEADS);
    float* Em   = alloc((size_t)NNODE * HID * LAYERS);   // n1 outputs, [N,256]
    float* Ed   = alloc((size_t)NNODE * HID * LAYERS);   // n2 outputs, [N,256]

    const int TPB = 256;
    const int GEMM_B   = NNODE / MROWS;                    // 125 blocks
    const int NH       = NNODE * HEADS;                    // 80000
    const int NH_B     = (NH + TPB - 1) / TPB;
    const int NHID_B   = (NNODE * HID + TPB - 1) / TPB;
    const int E_B      = (NE + TPB - 1) / TPB;
    const unsigned ORD_NEG_INF = ~0xFF800000u;             // ordered encoding of -inf

    // input projections (with relu)
    han_gemm_kernel<<<GEMM_B, TPB, 0, stream>>>(x1, lin1_w, lin1_b, h1, 1);
    han_gemm_kernel<<<GEMM_B, TPB, 0, stream>>>(x2, lin2_w, lin2_b, h2, 1);

    for (int l = 0; l < LAYERS; ++l) {
        const float* pw1 = (const float*)d_in[6 + l * 8 + 0];
        const float* pb1 = (const float*)d_in[6 + l * 8 + 1];
        const float* pw2 = (const float*)d_in[6 + l * 8 + 2];
        const float* pb2 = (const float*)d_in[6 + l * 8 + 3];
        const float* a_src12 = (const float*)d_in[6 + l * 8 + 4];
        const float* a_dst12 = (const float*)d_in[6 + l * 8 + 5];
        const float* a_src21 = (const float*)d_in[6 + l * 8 + 6];
        const float* a_dst21 = (const float*)d_in[6 + l * 8 + 7];

        han_gemm_kernel<<<GEMM_B, TPB, 0, stream>>>(h1, pw1, pb1, p1, 0);
        han_gemm_kernel<<<GEMM_B, TPB, 0, stream>>>(h2, pw2, pb2, p2, 0);

        // ---- edge type 1->2 : src=p1, dst=p2, result -> new h2 + Ed slice
        han_score_kernel<<<NH_B, TPB, 0, stream>>>(p1, a_src12, sA);
        han_score_kernel<<<NH_B, TPB, 0, stream>>>(p2, a_dst12, sB);
        han_fill_u32<<<NH_B, TPB, 0, stream>>>(amaxU, ORD_NEG_INF, NH);
        han_fill_u32<<<NH_B, TPB, 0, stream>>>((unsigned*)den, 0u, NH);
        han_fill_u32<<<NHID_B, TPB, 0, stream>>>((unsigned*)agg, 0u, NNODE * HID);
        han_edge_max_kernel<<<E_B, TPB, 0, stream>>>(e12, sA, sB, amaxU);
        han_amax_fin_kernel<<<NH_B, TPB, 0, stream>>>(amaxU, amaxF);
        han_edge_den_kernel<<<E_B, TPB, 0, stream>>>(e12, sA, sB, amaxF, den);
        han_edge_scatter_kernel<<<NE / 8, TPB, 0, stream>>>(e12, p1, sA, sB, amaxF, den, agg);
        han_relu_copy_kernel<<<NHID_B, TPB, 0, stream>>>(agg, h2, Ed, l);

        // ---- edge type 2->1 : src=p2, dst=p1, result -> new h1 + Em slice
        han_score_kernel<<<NH_B, TPB, 0, stream>>>(p2, a_src21, sA);
        han_score_kernel<<<NH_B, TPB, 0, stream>>>(p1, a_dst21, sB);
        han_fill_u32<<<NH_B, TPB, 0, stream>>>(amaxU, ORD_NEG_INF, NH);
        han_fill_u32<<<NH_B, TPB, 0, stream>>>((unsigned*)den, 0u, NH);
        han_fill_u32<<<NHID_B, TPB, 0, stream>>>((unsigned*)agg, 0u, NNODE * HID);
        han_edge_max_kernel<<<E_B, TPB, 0, stream>>>(e21, sA, sB, amaxU);
        han_amax_fin_kernel<<<NH_B, TPB, 0, stream>>>(amaxU, amaxF);
        han_edge_den_kernel<<<E_B, TPB, 0, stream>>>(e21, sA, sB, amaxF, den);
        han_edge_scatter_kernel<<<NE / 8, TPB, 0, stream>>>(e21, p2, sA, sB, amaxF, den, agg);
        han_relu_copy_kernel<<<NHID_B, TPB, 0, stream>>>(agg, h1, Em, l);
    }

    // pair scoring: out[p] = dot(Em[m[p]], Ed[d[p]])
    han_pair_kernel<<<NP / 8, TPB, 0, stream>>>(pidx, Em, Ed, out);
}
